// AtomTransformer_64441689309753
// MI455X (gfx1250) — compile-verified, matrix operand order checked by reading, not run on hardware
//
#include <hip/hip_runtime.h>

typedef __attribute__((ext_vector_type(16))) _Float16 v16h;
typedef __attribute__((ext_vector_type(8)))  _Float16 h8;
typedef __attribute__((ext_vector_type(8)))  float    v8f;

#define BDIM 4
#define NDIM 1024
#define RTOT 4096   // B*N
#define HNUM 4
#define DDIM 32
#define NPAD 1152   // 1024 + 2*64 halo for padded V-transposed layout

__device__ __forceinline__ float fast_sigmoid(float x) {
  return __builtin_amdgcn_rcpf(1.0f + __expf(-x));
}

__device__ __forceinline__ v8f wmma_f16(v16h a, v16h b, v8f c) {
  return __builtin_amdgcn_wmma_f32_16x16x32_f16(false, a, false, b, (short)0, c, false, false);
}

// A fragment: 16x32 f16, row-major src, leading dim ld (halves).
// Lane l holds row m = row0 + (l&15); hi = l>>4 selects K groups per ISA layout.
// Per-lane data = two contiguous 8-half (16B) chunks at k0 + hi*8 and +16.
__device__ __forceinline__ v16h frag_a(const _Float16* src, int ld, int row0, int k0) {
  int lane = threadIdx.x & 31;
  int m = row0 + (lane & 15);
  int hi = lane >> 4;
  const h8* p0 = (const h8*)(src + (size_t)m * ld + k0 + (hi << 3));
  union { v16h v; h8 h[2]; } u;
  u.h[0] = p0[0];
  u.h[1] = p0[2];  // +16 halves
  return u.v;
}

// B fragment from transposed storage: element (k, n) stored at src[n*ld + k].
// Lane l holds column n = col0 + (l&15); same K pattern as A -> two 16B loads.
__device__ __forceinline__ v16h frag_bt(const _Float16* src, int ld, int col0, int k0) {
  int lane = threadIdx.x & 31;
  int n = col0 + (lane & 15);
  int hi = lane >> 4;
  const h8* p0 = (const h8*)(src + (size_t)n * ld + k0 + (hi << 3));
  union { v16h v; h8 h[2]; } u;
  u.h[0] = p0[0];
  u.h[1] = p0[2];
  return u.v;
}

__device__ __forceinline__ v16h frag_bt_clamp(const _Float16* src, int ld, int col0, int k0, int maxn) {
  int lane = threadIdx.x & 31;
  int n = col0 + (lane & 15);
  n = n < 0 ? 0 : (n >= maxn ? maxn - 1 : n);
  int hi = lane >> 4;
  const h8* p0 = (const h8*)(src + (size_t)n * ld + k0 + (hi << 3));
  union { v16h v; h8 h[2]; } u;
  u.h[0] = p0[0];
  u.h[1] = p0[2];
  return u.v;
}

// ---------------- elementwise / prep kernels ----------------

__global__ void conv_kernel(const float* __restrict__ src, _Float16* __restrict__ dst, int n) {
  int i = blockIdx.x * blockDim.x + threadIdx.x;
  if (i < n) dst[i] = (_Float16)src[i];
}

// f32 [rows,cols] -> f16 transposed [cols][rows], optional per-row (K-index) scale fold.
__global__ void conv_t_kernel(const float* __restrict__ src, int rows, int cols,
                              _Float16* __restrict__ dst, const float* __restrict__ rowscale) {
  int i = blockIdx.x * blockDim.x + threadIdx.x;
  if (i >= rows * cols) return;
  int r = i / cols, c = i % cols;
  float v = src[i];
  if (rowscale) v *= rowscale[r];
  dst[(size_t)c * rows + r] = (_Float16)v;
}

// wbg[cp][h] = lnz_g[cp]*wb[cp][h]; cbias[h] = sum_cp lnz_b[cp]*wb[cp][h]
__global__ void wb_prep_kernel(const float* __restrict__ g, const float* __restrict__ b,
                               const float* __restrict__ wb, float* __restrict__ wbg,
                               float* __restrict__ cbias) {
  int i = threadIdx.x;
  if (i < 64) wbg[i] = g[i >> 2] * wb[i];
  if (i < 4) {
    float s = 0.f;
    for (int cp = 0; cp < 16; ++cp) s += b[cp] * wb[cp * 4 + i];
    cbias[i] = s;
  }
}

// Row LayerNorm (population var, eps=1e-5), f32 in -> f16 out. One wave per row.
__global__ void ln_rows_kernel(const float* __restrict__ src, _Float16* __restrict__ dst,
                               int rows, int cols) {
  int wave = threadIdx.x >> 5;
  int lane = threadIdx.x & 31;
  int row = blockIdx.x * (blockDim.x >> 5) + wave;
  if (row >= rows) return;
  const float* s = src + (size_t)row * cols;
  int per = cols >> 5;
  float x[4];
  float sum = 0.f;
#pragma unroll 4
  for (int i = 0; i < per; ++i) { x[i] = s[lane + (i << 5)]; sum += x[i]; }
#pragma unroll
  for (int m = 1; m < 32; m <<= 1) sum += __shfl_xor(sum, m, 32);
  float mean = sum / (float)cols;
  float sq = 0.f;
#pragma unroll 4
  for (int i = 0; i < per; ++i) { float d = x[i] - mean; sq += d * d; }
#pragma unroll
  for (int m = 1; m < 32; m <<= 1) sq += __shfl_xor(sq, m, 32);
  float inv = rsqrtf(sq / (float)cols + 1e-5f);
#pragma unroll 4
  for (int i = 0; i < per; ++i)
    dst[(size_t)row * cols + lane + (i << 5)] = (_Float16)((x[i] - mean) * inv);
}

// ---------- WMMA GEMM kernels: 1 wave = 16x64 of C (4 tiles, A-frag reuse) ----------

// ah = sigmoid(cn@WgT) * an + cn@WsT   (K=64, N=128)
__global__ void adaln_gemm_kernel(const _Float16* __restrict__ cn, const _Float16* __restrict__ wgT,
                                  const _Float16* __restrict__ wsT, const _Float16* __restrict__ an,
                                  _Float16* __restrict__ out, int rows) {
  int lane = threadIdx.x & 31;
  int task = blockIdx.x * (blockDim.x >> 5) + (threadIdx.x >> 5);
  if (task >= (rows >> 4) * 2) return;
  int mt = task >> 1, g = task & 1;
  v8f acc1[4] = {};
  v8f acc2[4] = {};
#pragma unroll
  for (int ks = 0; ks < 2; ++ks) {
    v16h a = frag_a(cn, 64, mt * 16, ks * 32);
#pragma unroll
    for (int t = 0; t < 4; ++t) {
      acc1[t] = wmma_f16(a, frag_bt(wgT, 64, g * 64 + t * 16, ks * 32), acc1[t]);
      acc2[t] = wmma_f16(a, frag_bt(wsT, 64, g * 64 + t * 16, ks * 32), acc2[t]);
    }
  }
  int hi = lane >> 4;
#pragma unroll
  for (int t = 0; t < 4; ++t)
#pragma unroll
    for (int v = 0; v < 8; ++v) {
      int r = mt * 16 + v + (hi << 3);
      int c = g * 64 + t * 16 + (lane & 15);
      float anv = (float)an[(size_t)r * 128 + c];
      out[(size_t)r * 128 + c] = (_Float16)(fast_sigmoid(acc1[t][v]) * anv + acc2[t][v]);
    }
}

// [rows,128] @ W[128,512] -> q (bias + 1/sqrt(D) folded), k in [B,H,N,D];
// v in transposed+padded [B,H,D,NPAD] (offset +64); g=sigmoid in [rows,128]
__global__ void qkvg_gemm_kernel(const _Float16* __restrict__ ah, const _Float16* __restrict__ wT,
                                 const float* __restrict__ bq, _Float16* __restrict__ qb,
                                 _Float16* __restrict__ kb, _Float16* __restrict__ vt,
                                 _Float16* __restrict__ gb, int rows) {
  int lane = threadIdx.x & 31;
  int task = blockIdx.x * (blockDim.x >> 5) + (threadIdx.x >> 5);
  if (task >= (rows >> 4) * 8) return;
  int mt = task >> 3, g = task & 7;
  v8f acc[4] = {};
#pragma unroll
  for (int ks = 0; ks < 4; ++ks) {
    v16h a = frag_a(ah, 128, mt * 16, ks * 32);
#pragma unroll
    for (int t = 0; t < 4; ++t)
      acc[t] = wmma_f16(a, frag_bt(wT, 128, g * 64 + t * 16, ks * 32), acc[t]);
  }
  int hi = lane >> 4;
#pragma unroll
  for (int t = 0; t < 4; ++t)
#pragma unroll
    for (int v = 0; v < 8; ++v) {
      int r = mt * 16 + v + (hi << 3);
      int C = g * 64 + t * 16 + (lane & 15);
      int seg = C >> 7, cc = C & 127;
      int hd = cc >> 5, dd = cc & 31;
      int b = r >> 10, n = r & 1023;
      float val = acc[t][v];
      if (seg == 0) {
        qb[((((size_t)b * HNUM + hd) * NDIM) + n) * DDIM + dd] =
            (_Float16)((val + bq[cc]) * 0.17677669529663687f);
      } else if (seg == 1) {
        kb[((((size_t)b * HNUM + hd) * NDIM) + n) * DDIM + dd] = (_Float16)val;
      } else if (seg == 2) {
        vt[(((size_t)b * HNUM + hd) * DDIM + dd) * NPAD + 64 + n] = (_Float16)val;
      } else {
        gb[(size_t)r * 128 + cc] = (_Float16)fast_sigmoid(val);
      }
    }
}

// out = sigmoid(c16 @ wT + bias)   (K=64, N=128, f32 out)
__global__ void sig_gemm_kernel(const _Float16* __restrict__ c16, const _Float16* __restrict__ wT,
                                const float* __restrict__ bias, float* __restrict__ out, int rows) {
  int lane = threadIdx.x & 31;
  int task = blockIdx.x * (blockDim.x >> 5) + (threadIdx.x >> 5);
  if (task >= (rows >> 4) * 2) return;
  int mt = task >> 1, g = task & 1;
  v8f acc[4] = {};
#pragma unroll
  for (int ks = 0; ks < 2; ++ks) {
    v16h a = frag_a(c16, 64, mt * 16, ks * 32);
#pragma unroll
    for (int t = 0; t < 4; ++t)
      acc[t] = wmma_f16(a, frag_bt(wT, 64, g * 64 + t * 16, ks * 32), acc[t]);
  }
  int hi = lane >> 4;
#pragma unroll
  for (int t = 0; t < 4; ++t)
#pragma unroll
    for (int v = 0; v < 8; ++v) {
      int r = mt * 16 + v + (hi << 3);
      int c = g * 64 + t * 16 + (lane & 15);
      out[(size_t)r * 128 + c] = fast_sigmoid(acc[t][v] + bias[c]);
    }
}

// attn_out = sigA * (go @ woT)   (K=128, N=128, f32 out)
__global__ void wo_gemm_kernel(const _Float16* __restrict__ go, const _Float16* __restrict__ woT,
                               const float* __restrict__ sigA, float* __restrict__ out, int rows) {
  int lane = threadIdx.x & 31;
  int task = blockIdx.x * (blockDim.x >> 5) + (threadIdx.x >> 5);
  if (task >= (rows >> 4) * 2) return;
  int mt = task >> 1, g = task & 1;
  v8f acc[4] = {};
#pragma unroll
  for (int ks = 0; ks < 4; ++ks) {
    v16h a = frag_a(go, 128, mt * 16, ks * 32);
#pragma unroll
    for (int t = 0; t < 4; ++t)
      acc[t] = wmma_f16(a, frag_bt(woT, 128, g * 64 + t * 16, ks * 32), acc[t]);
  }
  int hi = lane >> 4;
#pragma unroll
  for (int t = 0; t < 4; ++t)
#pragma unroll
    for (int v = 0; v < 8; ++v) {
      int r = mt * 16 + v + (hi << 3);
      int c = g * 64 + t * 16 + (lane & 15);
      size_t idx = (size_t)r * 128 + c;
      out[idx] = sigA[idx] * acc[t][v];
    }
}

// bm = silu(t@w1) * (t@w2), W12 transposed [512][128], out cols 0..255
__global__ void bm_gemm_kernel(const _Float16* __restrict__ t16, const _Float16* __restrict__ w12T,
                               _Float16* __restrict__ bm, int rows) {
  int lane = threadIdx.x & 31;
  int task = blockIdx.x * (blockDim.x >> 5) + (threadIdx.x >> 5);
  if (task >= (rows >> 4) * 4) return;
  int mt = task >> 2, g = task & 3;
  v8f acc1[4] = {};
  v8f acc2[4] = {};
#pragma unroll
  for (int ks = 0; ks < 4; ++ks) {
    v16h a = frag_a(t16, 128, mt * 16, ks * 32);
#pragma unroll
    for (int t = 0; t < 4; ++t) {
      acc1[t] = wmma_f16(a, frag_bt(w12T, 128, g * 64 + t * 16, ks * 32), acc1[t]);
      acc2[t] = wmma_f16(a, frag_bt(w12T, 128, 256 + g * 64 + t * 16, ks * 32), acc2[t]);
    }
  }
  int hi = lane >> 4;
#pragma unroll
  for (int t = 0; t < 4; ++t)
#pragma unroll
    for (int v = 0; v < 8; ++v) {
      int r = mt * 16 + v + (hi << 3);
      int c = g * 64 + t * 16 + (lane & 15);
      float x1 = acc1[t][v];
      bm[(size_t)r * 256 + c] = (_Float16)((x1 * fast_sigmoid(x1)) * acc2[t][v]);
    }
}

// dst = sigT * (bm @ woutT) + attn_out   (K=256, N=128)
__global__ void final_gemm_kernel(const _Float16* __restrict__ bm, const _Float16* __restrict__ woutT,
                                  const float* __restrict__ sigT, const float* __restrict__ attn_out,
                                  float* __restrict__ dst, int rows) {
  int lane = threadIdx.x & 31;
  int task = blockIdx.x * (blockDim.x >> 5) + (threadIdx.x >> 5);
  if (task >= (rows >> 4) * 2) return;
  int mt = task >> 1, g = task & 1;
  v8f acc[4] = {};
#pragma unroll
  for (int ks = 0; ks < 8; ++ks) {
    v16h a = frag_a(bm, 256, mt * 16, ks * 32);
#pragma unroll
    for (int t = 0; t < 4; ++t)
      acc[t] = wmma_f16(a, frag_bt(woutT, 256, g * 64 + t * 16, ks * 32), acc[t]);
  }
  int hi = lane >> 4;
#pragma unroll
  for (int t = 0; t < 4; ++t)
#pragma unroll
    for (int v = 0; v < 8; ++v) {
      int r = mt * 16 + v + (hi << 3);
      int c = g * 64 + t * 16 + (lane & 15);
      size_t idx = (size_t)r * 128 + c;
      dst[idx] = sigT[idx] * acc[t][v] + attn_out[idx];
    }
}

// ---------------- fused local attention with pair bias ----------------
// 1 workgroup (128 thr = 4 waves) per (batch, query-block j). Wave w handles head w.
// Query window [32j,32j+31], key window [32j-48, 32j+79]. Pair-bias LN+proj computed
// cooperatively once into LDS (shared by all 4 heads). V read from padded transposed
// layout so the PV B-fragment is two 16B loads per lane; halo columns have P==0.
__global__ void attn_kernel(const _Float16* __restrict__ qb, const _Float16* __restrict__ kb,
                            const _Float16* __restrict__ vt, const _Float16* __restrict__ gb,
                            const float* __restrict__ p, const float* __restrict__ wbg,
                            const float* __restrict__ cbias, _Float16* __restrict__ go) {
  int bj = blockIdx.x;
  int b = bj >> 5;
  int j = bj & 31;
  int q0 = j * 32;
  int w0 = q0 - 48;
  int tid = threadIdx.x;
  int wave = tid >> 5;
  int lane = tid & 31;
  int hi = lane >> 4;

  __shared__ __align__(16) _Float16 biasSh[32][128][4];  // 32 KB
  __shared__ __align__(16) _Float16 Psh[4][32][128];     // 32 KB

  // ---- pair-bias pre-pass: LN over 16 channels + 16x4 projection ----
  for (int pair = tid; pair < 32 * 128; pair += 128) {
    int ql = pair >> 7;
    int kl = pair & 127;
    int kg = w0 + kl;
    float v4[4] = {cbias[0], cbias[1], cbias[2], cbias[3]};
    if (kg >= 0 && kg < NDIM) {
      const float* pr = p + (((size_t)(b * NDIM + q0 + ql) * NDIM + kg) << 4);
      float x[16];
      float s = 0.f;
#pragma unroll
      for (int i = 0; i < 16; ++i) { x[i] = pr[i]; s += x[i]; }
      float mean = s * 0.0625f;
      float ss = 0.f;
#pragma unroll
      for (int i = 0; i < 16; ++i) { float d0 = x[i] - mean; ss += d0 * d0; }
      float inv = rsqrtf(ss * 0.0625f + 1e-5f);
#pragma unroll
      for (int i = 0; i < 16; ++i) {
        float ln = (x[i] - mean) * inv;
        v4[0] += ln * wbg[i * 4 + 0];
        v4[1] += ln * wbg[i * 4 + 1];
        v4[2] += ln * wbg[i * 4 + 2];
        v4[3] += ln * wbg[i * 4 + 3];
      }
    }
#pragma unroll
    for (int h = 0; h < 4; ++h) biasSh[ql][kl][h] = (_Float16)v4[h];
  }
  __syncthreads();

  int hd = wave;
  const _Float16* Q = qb + (size_t)(b * HNUM + hd) * NDIM * DDIM;
  const _Float16* K = kb + (size_t)(b * HNUM + hd) * NDIM * DDIM;
  const _Float16* Vt = vt + (size_t)(b * HNUM + hd) * DDIM * NPAD;

  // ---- scores + softmax + P (two 16-row tiles) ----
  for (int mt = 0; mt < 2; ++mt) {
    v8f S[8];
    v16h aq = frag_a(Q, DDIM, q0 + mt * 16, 0);  // 16x32 (full D)
#pragma unroll
    for (int nt = 0; nt < 8; ++nt) {
      v8f z = {};
      S[nt] = wmma_f16(aq, frag_bt_clamp(K, DDIM, w0 + nt * 16, 0, NDIM), z);
    }
    // add pair bias + edge mask
#pragma unroll
    for (int nt = 0; nt < 8; ++nt) {
#pragma unroll
      for (int v = 0; v < 8; ++v) {
        int ql = mt * 16 + v + (hi << 3);
        int kl = nt * 16 + (lane & 15);
        int kg = w0 + kl;
        S[nt][v] = (kg < 0 || kg >= NDIM) ? -1e30f : (S[nt][v] + (float)biasSh[ql][kl][hd]);
      }
    }
    // row softmax: rows live in VGPR index v within each lane-half; cols across lanes 0..15
    float rmax[8], rinv[8];
#pragma unroll
    for (int v = 0; v < 8; ++v) {
      float m0 = -1e30f;
#pragma unroll
      for (int nt = 0; nt < 8; ++nt) m0 = fmaxf(m0, S[nt][v]);
#pragma unroll
      for (int msk = 1; msk <= 8; msk <<= 1) m0 = fmaxf(m0, __shfl_xor(m0, msk, 32));
      rmax[v] = m0;
    }
#pragma unroll
    for (int v = 0; v < 8; ++v) {
      float s0 = 0.f;
#pragma unroll
      for (int nt = 0; nt < 8; ++nt) {
        float e = __expf(S[nt][v] - rmax[v]);
        S[nt][v] = e;
        s0 += e;
      }
#pragma unroll
      for (int msk = 1; msk <= 8; msk <<= 1) s0 += __shfl_xor(s0, msk, 32);
      rinv[v] = __builtin_amdgcn_rcpf(s0);
    }
#pragma unroll
    for (int nt = 0; nt < 8; ++nt)
#pragma unroll
      for (int v = 0; v < 8; ++v)
        Psh[wave][mt * 16 + v + (hi << 3)][nt * 16 + (lane & 15)] = (_Float16)(S[nt][v] * rinv[v]);
  }

  // ---- O = P @ V_window (all fragments vectorized), gate, store go ----
  const _Float16* Pw = &Psh[wave][0][0];
  const _Float16* Vw = Vt + 64 + w0;  // window base inside padded row
  for (int mt = 0; mt < 2; ++mt) {
    for (int dt = 0; dt < 2; ++dt) {
      v8f acc = {};
#pragma unroll
      for (int ks = 0; ks < 4; ++ks)
        acc = wmma_f16(frag_a(Pw, 128, mt * 16, ks * 32),
                       frag_bt(Vw, NPAD, dt * 16, ks * 32), acc);
#pragma unroll
      for (int v = 0; v < 8; ++v) {
        int n = q0 + mt * 16 + v + (hi << 3);
        int c = hd * DDIM + dt * 16 + (lane & 15);
        size_t idx = (size_t)(b * NDIM + n) * 128 + c;
        go[idx] = (_Float16)(acc[v] * (float)gb[idx]);
      }
    }
  }
}

// ---------------- host orchestration ----------------

extern "C" void kernel_launch(void* const* d_in, const int* in_sizes, int n_in,
                              void* d_out, int out_size, void* d_ws, size_t ws_size,
                              hipStream_t stream) {
  (void)in_sizes; (void)n_in; (void)out_size; (void)ws_size;
  const float* q_in          = (const float*)d_in[0];
  const float* c_in          = (const float*)d_in[1];
  const float* p_in          = (const float*)d_in[2];
  const float* attn_s_gamma  = (const float*)d_in[3];
  const float* attn_gate_w   = (const float*)d_in[4];
  const float* attn_skip_w   = (const float*)d_in[5];
  const float* attn_wq       = (const float*)d_in[6];
  const float* attn_bq       = (const float*)d_in[7];
  const float* attn_wk       = (const float*)d_in[8];
  const float* attn_wv       = (const float*)d_in[9];
  const float* attn_lnz_g    = (const float*)d_in[10];
  const float* attn_lnz_b    = (const float*)d_in[11];
  const float* attn_wb       = (const float*)d_in[12];
  const float* attn_wgate    = (const float*)d_in[13];
  const float* attn_wo       = (const float*)d_in[14];
  const float* attn_ws_out   = (const float*)d_in[15];
  const float* attn_bs_out   = (const float*)d_in[16];
  const float* trans_s_gamma = (const float*)d_in[17];
  const float* trans_gate_w  = (const float*)d_in[18];
  const float* trans_skip_w  = (const float*)d_in[19];
  const float* trans_w1      = (const float*)d_in[20];
  const float* trans_w2      = (const float*)d_in[21];
  const float* trans_wout    = (const float*)d_in[22];
  const float* trans_ws_out  = (const float*)d_in[23];
  const float* trans_bs_out  = (const float*)d_in[24];

  const int R = RTOT;
  size_t off = 0;
  auto alloc = [&](size_t bytes) -> void* {
    void* ptr = (char*)d_ws + off;
    off += (bytes + 255) & ~(size_t)255;
    return ptr;
  };
  _Float16* c16   = (_Float16*)alloc((size_t)R * 64 * 2);
  _Float16* cn16  = (_Float16*)alloc((size_t)R * 64 * 2);
  _Float16* an16  = (_Float16*)alloc((size_t)R * 128 * 2);
  _Float16* x16   = (_Float16*)alloc((size_t)R * 128 * 2);
  _Float16* qh16  = (_Float16*)alloc((size_t)R * 128 * 2);
  _Float16* kh16  = (_Float16*)alloc((size_t)R * 128 * 2);
  _Float16* vt16  = (_Float16*)alloc((size_t)BDIM * HNUM * DDIM * NPAD * 2);
  _Float16* g16   = (_Float16*)alloc((size_t)R * 128 * 2);
  _Float16* go16  = (_Float16*)alloc((size_t)R * 128 * 2);
  _Float16* bm16  = (_Float16*)alloc((size_t)R * 256 * 2);
  float*    sigA  = (float*)alloc((size_t)R * 128 * 4);
  float*    sigT  = (float*)alloc((size_t)R * 128 * 4);
  float*    aout  = (float*)alloc((size_t)R * 128 * 4);
  float*    a_buf = (float*)alloc((size_t)R * 128 * 4);
  _Float16* WG    = (_Float16*)alloc(128 * 64 * 2);
  _Float16* WS    = (_Float16*)alloc(128 * 64 * 2);
  _Float16* WQKVG = (_Float16*)alloc(512 * 128 * 2);
  _Float16* WO    = (_Float16*)alloc(128 * 128 * 2);
  _Float16* WSA   = (_Float16*)alloc(128 * 64 * 2);
  _Float16* WST   = (_Float16*)alloc(128 * 64 * 2);
  _Float16* TG    = (_Float16*)alloc(128 * 64 * 2);
  _Float16* TS    = (_Float16*)alloc(128 * 64 * 2);
  _Float16* W12   = (_Float16*)alloc(512 * 128 * 2);
  _Float16* WOUT  = (_Float16*)alloc(128 * 256 * 2);
  float*    wbg   = (float*)alloc(64 * 4);
  float*    cbias = (float*)alloc(4 * 4);

  // once: c in f16 (raw + LN'd)
  conv_kernel<<<(R * 64 + 255) / 256, 256, 0, stream>>>(c_in, c16, R * 64);
  ln_rows_kernel<<<R / 8, 256, 0, stream>>>(c_in, cn16, R, 64);

  const float* a_src = q_in;
  for (int i = 0; i < 3; ++i) {
    // weight prep (transposed f16; gamma folded into AdaLN weights)
    conv_t_kernel<<<32, 256, 0, stream>>>(attn_gate_w + (size_t)i * 8192, 64, 128, WG, attn_s_gamma + i * 64);
    conv_t_kernel<<<32, 256, 0, stream>>>(attn_skip_w + (size_t)i * 8192, 64, 128, WS, attn_s_gamma + i * 64);
    conv_t_kernel<<<64, 256, 0, stream>>>(attn_wq + (size_t)i * 16384, 128, 128, WQKVG + 0 * 16384, nullptr);
    conv_t_kernel<<<64, 256, 0, stream>>>(attn_wk + (size_t)i * 16384, 128, 128, WQKVG + 1 * 16384, nullptr);
    conv_t_kernel<<<64, 256, 0, stream>>>(attn_wv + (size_t)i * 16384, 128, 128, WQKVG + 2 * 16384, nullptr);
    conv_t_kernel<<<64, 256, 0, stream>>>(attn_wgate + (size_t)i * 16384, 128, 128, WQKVG + 3 * 16384, nullptr);
    conv_t_kernel<<<64, 256, 0, stream>>>(attn_wo + (size_t)i * 16384, 128, 128, WO, nullptr);
    conv_t_kernel<<<32, 256, 0, stream>>>(attn_ws_out + (size_t)i * 8192, 64, 128, WSA, nullptr);
    conv_t_kernel<<<32, 256, 0, stream>>>(trans_gate_w + (size_t)i * 8192, 64, 128, TG, trans_s_gamma + i * 64);
    conv_t_kernel<<<32, 256, 0, stream>>>(trans_skip_w + (size_t)i * 8192, 64, 128, TS, trans_s_gamma + i * 64);
    conv_t_kernel<<<128, 256, 0, stream>>>(trans_w1 + (size_t)i * 32768, 128, 256, W12 + 0 * 32768, nullptr);
    conv_t_kernel<<<128, 256, 0, stream>>>(trans_w2 + (size_t)i * 32768, 128, 256, W12 + 1 * 32768, nullptr);
    conv_t_kernel<<<128, 256, 0, stream>>>(trans_wout + (size_t)i * 32768, 256, 128, WOUT, nullptr);
    conv_t_kernel<<<32, 256, 0, stream>>>(trans_ws_out + (size_t)i * 8192, 64, 128, WST, nullptr);
    wb_prep_kernel<<<1, 64, 0, stream>>>(attn_lnz_g + i * 16, attn_lnz_b + i * 16, attn_wb + i * 64, wbg, cbias);

    // LN of block input a (shared by attention AdaLN and transition AdaLN)
    ln_rows_kernel<<<R / 8, 256, 0, stream>>>(a_src, an16, R, 128);

    // attention path
    adaln_gemm_kernel<<<128, 128, 0, stream>>>(cn16, WG, WS, an16, x16, R);
    qkvg_gemm_kernel<<<512, 128, 0, stream>>>(x16, WQKVG, attn_bq + i * 128, qh16, kh16, vt16, g16, R);
    sig_gemm_kernel<<<128, 128, 0, stream>>>(c16, WSA, attn_bs_out + i * 128, sigA, R);
    sig_gemm_kernel<<<128, 128, 0, stream>>>(c16, WST, trans_bs_out + i * 128, sigT, R);
    attn_kernel<<<BDIM * 32, 128, 0, stream>>>(qh16, kh16, vt16, g16, p_in, wbg, cbias, go16);
    wo_gemm_kernel<<<128, 128, 0, stream>>>(go16, WO, sigA, aout, R);

    // transition path
    adaln_gemm_kernel<<<128, 128, 0, stream>>>(cn16, TG, TS, an16, x16, R);
    bm_gemm_kernel<<<256, 128, 0, stream>>>(x16, W12, bm16, R);
    float* dst = (i == 2) ? (float*)d_out : a_buf;
    final_gemm_kernel<<<128, 128, 0, stream>>>(bm16, WOUT, sigT, aout, dst, R);
    a_src = a_buf;
  }
}
